// CustomGPS_80109730005376
// MI455X (gfx1250) — matrix operand order repaired and checked by reading, not compiled, not run on hardware
//
#include <hip/hip_runtime.h>
#include <hip/hip_bf16.h>

// ---------------------------------------------------------------------------
// Problem constants (from the reference)
// ---------------------------------------------------------------------------
#define NN   32768      // nodes
#define EE   524288     // edges
#define CC   256        // channels
#define HH   8          // heads
#define BB   128        // graphs
#define PER  256        // nodes per graph
#define HD   32         // head dim
#define NC   10         // classes

typedef __attribute__((ext_vector_type(16))) _Float16 v16h;
typedef __attribute__((ext_vector_type(8)))  float    v8f;
typedef __attribute__((ext_vector_type(4)))  unsigned int u32x4;
typedef __attribute__((ext_vector_type(8)))  int      i32x8;
typedef __attribute__((ext_vector_type(4)))  int      i32x4;

#if defined(__has_builtin)
#  if __has_builtin(__builtin_amdgcn_tensor_load_to_lds)
#    define HAVE_TDM 1
#  else
#    define HAVE_TDM 0
#  endif
#else
#  define HAVE_TDM 0
#endif

// ---------------------------------------------------------------------------
// Generic f16-WMMA GEMM:  Y[M,Nout] = X[M,K] * (TRANSB ? W[Nout,K]^T : W[K,Nout])
//                         (+ bias) (+ relu).  fp32 inputs converted to f16,
//                         fp32 accumulate via v_wmma_f32_16x16x32_f16.
// The 16-column B panel (full K, <=32KB) is DMA'd into LDS once per block by
// the Tensor Data Mover; all 8 waves build B fragments from LDS.
// One wave computes one 16x16 tile; block = 8 waves = 8 M-tiles.
// ---------------------------------------------------------------------------
template<bool TRANSB, bool BIAS, bool RELU>
__global__ __launch_bounds__(256)
void gemm_wmma(const float* __restrict__ X, const float* __restrict__ W,
               const float* __restrict__ bias, float* __restrict__ Y,
               int M, int K, int Nout)
{
    __shared__ float BsF[512 * 16];          // K<=512 columns panel, fp32, 32KB

    const int lane  = threadIdx.x & 31;
    const int wave  = threadIdx.x >> 5;
    const int tileM = blockIdx.x * 8 + wave;
    const int n0    = blockIdx.y * 16;

#if HAVE_TDM
    // ---- TDM stage: one 2-D descriptor per workgroup (wave 0 issues) ----
    if (threadIdx.x < 32) {
        unsigned long long gaddr;
        unsigned int td0, td1, tile0, tile1;
        unsigned long long stride0;
        if (TRANSB) {   // panel = W[n0..n0+15][0..K)  -> LDS[col][k]
            gaddr = (unsigned long long)(const void*)(W + (size_t)n0 * K);
            td0 = (unsigned int)K; td1 = 16;
            tile0 = (unsigned int)K; tile1 = 16;
            stride0 = (unsigned long long)K;
        } else {        // panel = W[0..K)[n0..n0+15]  -> LDS[k][col]
            gaddr = (unsigned long long)(const void*)(W + n0);
            td0 = 16; td1 = (unsigned int)K;
            tile0 = 16; tile1 = (unsigned int)K;
            stride0 = (unsigned long long)Nout;
        }
        const unsigned int lds = (unsigned int)(size_t)(void*)BsF;
        u32x4 g0;
        g0[0] = 1u;                                           // count=1, user D#
        g0[1] = lds;                                          // lds_addr
        g0[2] = (unsigned int)(gaddr & 0xFFFFFFFFull);        // global_addr[31:0]
        g0[3] = (unsigned int)((gaddr >> 32) & 0x1FFFFFFull)  // global_addr[56:32]
              | (2u << 30);                                   // type = 2 ("image")
        i32x8 g1;
        g1[0] = (int)(2u << 16);                              // data_size = 4B
        g1[1] = (int)((td0 & 0xFFFFu) << 16);                 // tensor_dim0[15:0]
        g1[2] = (int)((td0 >> 16) | ((td1 & 0xFFFFu) << 16)); // td0[31:16] | td1[15:0]
        g1[3] = (int)((td1 >> 16) | (tile0 << 16));           // td1[31:16] | tile_dim0
        g1[4] = (int)(tile1 & 0xFFFFu);                       // tile_dim1 (tile_dim2=0)
        g1[5] = (int)(stride0 & 0xFFFFFFFFull);               // dim0_stride[31:0]
        g1[6] = (int)((stride0 >> 32) & 0xFFFFull);           // dim0_stride[47:32]
        g1[7] = 0;
        const i32x4 z4 = {0, 0, 0, 0};                        // groups 2/3: 2-D tile
        const i32x8 z8 = {0, 0, 0, 0, 0, 0, 0, 0};
        __builtin_amdgcn_tensor_load_to_lds(g0, g1, z4, z4, z8, 0);
        __builtin_amdgcn_s_wait_tensorcnt(0);
    }
#else
    // ---- fallback: cooperative staged load ----
    if (TRANSB) {
        for (int i = threadIdx.x; i < 16 * K; i += 256) {
            int c = i / K, k = i - c * K;
            BsF[i] = W[(size_t)(n0 + c) * K + k];
        }
    } else {
        for (int i = threadIdx.x; i < K * 16; i += 256) {
            int k = i >> 4, n = i & 15;
            BsF[i] = W[(size_t)k * Nout + n0 + n];
        }
    }
#endif
    __syncthreads();

    const int  lh = lane & 15;
    const bool hi = lane >= 16;

    v8f acc = {};
    const int m0   = tileM * 16;
    const int arow = m0 + lh;
    const float* Xr = X + (size_t)arow * K;
    const int aoff = hi ? 16 : 0;  // A: lane<16 -> K {0..15}; else {16..31}

    for (int k0 = 0; k0 < K; k0 += 32) {
        v16h a;
#pragma unroll
        for (int e = 0; e < 8; ++e) {
            a[e]     = (_Float16)Xr[k0 + aoff + e];
            a[e + 8] = (_Float16)Xr[k0 + aoff + 8 + e];
        }
        const int kb = k0 + (hi ? 16 : 0);   // B: lane-half selects K-half
        v16h b;
        if (TRANSB) {
            const float* Br = BsF + (size_t)lh * K + kb;     // LDS[col][k]
#pragma unroll
            for (int e = 0; e < 16; ++e) b[e] = (_Float16)Br[e];
        } else {
            const float* Br = BsF + (size_t)kb * 16 + lh;    // LDS[k][col]
#pragma unroll
            for (int e = 0; e < 16; ++e) b[e] = (_Float16)Br[(size_t)e * 16];
        }
        acc = __builtin_amdgcn_wmma_f32_16x16x32_f16(false, a, false, b,
                                                     (short)0, acc, false, false);
    }

    const int rbase = m0 + (hi ? 8 : 0);   // C/D layout: vgpr r -> row r (+8 hi half)
    const int col   = n0 + lh;
    const float bv  = BIAS ? bias[col] : 0.0f;
#pragma unroll
    for (int r = 0; r < 8; ++r) {
        float v = acc[r] + bv;
        if (RELU) v = fmaxf(v, 0.0f);
        Y[(size_t)(rbase + r) * Nout + col] = v;
    }
}

// ---------------------------------------------------------------------------
// GCN helpers
// ---------------------------------------------------------------------------
__global__ void deg_init(float* deg, int n) {
    int i = blockIdx.x * blockDim.x + threadIdx.x;
    if (i < n) deg[i] = 1.0f;                          // self-loop
}
__global__ void deg_accum(const int* __restrict__ col, float* deg, int e_cnt) {
    int e = blockIdx.x * blockDim.x + threadIdx.x;
    if (e < e_cnt) atomicAdd(&deg[col[e]], 1.0f);
}
__global__ void deg_rsqrt(float* deg, int n) {
    int i = blockIdx.x * blockDim.x + threadIdx.x;
    if (i < n) deg[i] = rsqrtf(deg[i]);                // deg >= 1 always
}
// agg[i][c] = dinv[i]^2 * h[i][c]   (self-loop term)
__global__ void agg_init(const float* __restrict__ h, const float* __restrict__ dinv,
                         float* __restrict__ agg, int n) {
    int i = blockIdx.x * blockDim.x + threadIdx.x;
    if (i < n) {
        int r = i >> 8;
        float d = dinv[r];
        agg[i] = d * d * h[i];
    }
}
// scatter-add: one thread = (edge, 4-channel chunk)
__global__ __launch_bounds__(256)
void agg_edges(const int* __restrict__ ei, const float* __restrict__ h,
               const float* __restrict__ dinv, float* __restrict__ agg, int e_cnt) {
    int idx = blockIdx.x * 256 + threadIdx.x;
    int e   = idx >> 6;
    int c4  = (idx & 63) << 2;
    if (e < e_cnt) {
        int r  = ei[e];
        int cl = ei[e_cnt + e];
        __builtin_prefetch(h + (size_t)r * CC + c4, 0, 3);   // global_prefetch_b8
        float nrm = dinv[r] * dinv[cl];
        const float4 hv = *(const float4*)(h + (size_t)r * CC + c4);
        float* o = agg + (size_t)cl * CC + c4;
        atomicAdd(o + 0, nrm * hv.x);
        atomicAdd(o + 1, nrm * hv.y);
        atomicAdd(o + 2, nrm * hv.z);
        atomicAdd(o + 3, nrm * hv.w);
    }
}

// ---------------------------------------------------------------------------
// BatchNorm (training mode over node axis). val = A (+bias[c]) (+Badd)
// stats: per-channel sum / sumsq -> atomics into sums[0..255]/sums[256..511]
// ---------------------------------------------------------------------------
__global__ __launch_bounds__(256)
void bn_stats(const float* __restrict__ A, const float* __restrict__ Badd,
              const float* __restrict__ bias, float* __restrict__ sums) {
    int c  = threadIdx.x;
    int r0 = blockIdx.x * 128;
    float bv = bias ? bias[c] : 0.0f;
    float s = 0.0f, q = 0.0f;
    for (int r = r0; r < r0 + 128; ++r) {
        float v = A[(size_t)r * CC + c] + bv;
        if (Badd) v += Badd[(size_t)r * CC + c];
        s += v; q += v * v;
    }
    atomicAdd(&sums[c], s);
    atomicAdd(&sums[CC + c], q);
}
__global__ void bn_apply(const float* __restrict__ A, const float* __restrict__ Badd,
                         const float* __restrict__ bias, const float* __restrict__ sums,
                         const float* __restrict__ g, const float* __restrict__ beta,
                         const float* __restrict__ addAfter, float* __restrict__ Y, int n) {
    int i = blockIdx.x * blockDim.x + threadIdx.x;
    if (i >= n) return;
    int c = i & (CC - 1);
    float v = A[i];
    if (bias) v += bias[c];
    if (Badd) v += Badd[i];
    const float invN = 1.0f / (float)NN;
    float mu  = sums[c] * invN;
    float var = sums[CC + c] * invN - mu * mu;
    float o = (v - mu) * rsqrtf(var + 1e-5f) * g[c] + beta[c];
    if (addAfter) o += addAfter[i];
    Y[i] = o;
}

// ---------------------------------------------------------------------------
// Dense MHSA: one block per (64-row chunk, head, graph). mask is all-true.
// Phase1: S = (Q K^T)/sqrt(HD) via WMMA (K-dim = HD = 32 -> one WMMA/tile),
//         f16 scores in LDS. Phase2: row softmax (f32). Phase3: O = P V via
//         WMMA with P read from LDS.
// ---------------------------------------------------------------------------
__global__ __launch_bounds__(256)
void attn_kernel(const float* __restrict__ qkv, float* __restrict__ attn_out)
{
    __shared__ _Float16 S[64][PER + 8];
    const int chunk = blockIdx.x;   // 0..3  (64-row chunks)
    const int head  = blockIdx.y;   // 0..7
    const int b     = blockIdx.z;   // 0..127
    const int lane  = threadIdx.x & 31;
    const int wave  = threadIdx.x >> 5;
    const int  lh   = lane & 15;
    const bool hi   = lane >= 16;
    const float scale = 0.17677669529663687f;   // HD^-0.5

    const float* base = qkv + (size_t)b * PER * (3 * CC);

    // ---- Phase 1: scores -> LDS (f16) ----
    {
        const int rt    = wave >> 1;              // row tile 0..3
        const int chalf = wave & 1;               // column half
        const int i     = chunk * 64 + rt * 16 + lh;
        const float* qr = base + (size_t)i * (3 * CC) + head * HD;
        v16h a;
        const int aoff = hi ? 16 : 0;
#pragma unroll
        for (int e = 0; e < 8; ++e) {
            a[e]     = (_Float16)qr[aoff + e];
            a[e + 8] = (_Float16)qr[aoff + 8 + e];
        }
        for (int ct = chalf * 8; ct < chalf * 8 + 8; ++ct) {
            const int j     = ct * 16 + lh;
            const float* kr = base + (size_t)j * (3 * CC) + CC + head * HD + (hi ? 16 : 0);
            v16h bb;
#pragma unroll
            for (int e = 0; e < 16; ++e) bb[e] = (_Float16)kr[e];
            v8f c = {};
            c = __builtin_amdgcn_wmma_f32_16x16x32_f16(false, a, false, bb,
                                                       (short)0, c, false, false);
            const int rl = rt * 16 + (hi ? 8 : 0);
#pragma unroll
            for (int r = 0; r < 8; ++r)
                S[rl + r][ct * 16 + lh] = (_Float16)(c[r] * scale);
        }
    }
    __syncthreads();

    // ---- Phase 2: softmax over 256 cols; threads 0..63 own one row each ----
    if (threadIdx.x < 64) {
        const int row = threadIdx.x;
        float m = -1e30f;
        for (int j = 0; j < PER; ++j) m = fmaxf(m, (float)S[row][j]);
        float sum = 0.0f;
        for (int j = 0; j < PER; ++j) sum += __expf((float)S[row][j] - m);
        float inv = 1.0f / sum;
        for (int j = 0; j < PER; ++j)
            S[row][j] = (_Float16)(__expf((float)S[row][j] - m) * inv);
    }
    __syncthreads();

    // ---- Phase 3: O = P @ V ----
    {
        const int rt = wave >> 1;      // row tile 0..3
        const int dt = wave & 1;       // d-tile (HD=32 -> 2 tiles of 16)
        v8f c = {};
        const int rl = rt * 16 + lh;
        for (int kc = 0; kc < PER; kc += 32) {
            v16h a;
            const int kb = kc + (hi ? 16 : 0);
#pragma unroll
            for (int e = 0; e < 8; ++e) {
                a[e]     = S[rl][kb + e];
                a[e + 8] = S[rl][kb + 8 + e];
            }
            const float* vr = base + (size_t)(kc + (hi ? 16 : 0)) * (3 * CC)
                            + 2 * CC + head * HD + dt * 16 + lh;
            v16h bb;
#pragma unroll
            for (int e = 0; e < 16; ++e) bb[e] = (_Float16)vr[(size_t)e * (3 * CC)];
            c = __builtin_amdgcn_wmma_f32_16x16x32_f16(false, a, false, bb,
                                                       (short)0, c, false, false);
        }
        const int orow = b * PER + chunk * 64 + rt * 16 + (hi ? 8 : 0);
        const int ocol = head * HD + dt * 16 + lh;
#pragma unroll
        for (int r = 0; r < 8; ++r)
            attn_out[(size_t)(orow + r) * CC + ocol] = c[r];
    }
}

// ---------------------------------------------------------------------------
// segment_max pool (contiguous 256-row groups) + classifier with log_softmax
// ---------------------------------------------------------------------------
__global__ __launch_bounds__(256)
void pool_max(const float* __restrict__ X, float* __restrict__ pooled) {
    int b = blockIdx.x, c = threadIdx.x;
    const float* p = X + (size_t)b * PER * CC + c;
    float m = -1e30f;
    for (int r = 0; r < PER; ++r) m = fmaxf(m, p[(size_t)r * CC]);
    pooled[b * CC + c] = m;
}
__global__ __launch_bounds__(128)
void classify(const float* __restrict__ pooled, const float* __restrict__ fcw,
              const float* __restrict__ fcb, float* __restrict__ out) {
    int b = threadIdx.x;
    float lg[NC];
#pragma unroll
    for (int j = 0; j < NC; ++j) lg[j] = fcb[j];
    const float* pr = pooled + b * CC;
    for (int k = 0; k < CC; ++k) {
        float pv = pr[k];
#pragma unroll
        for (int j = 0; j < NC; ++j) lg[j] += pv * fcw[k * NC + j];
    }
    float m = lg[0];
#pragma unroll
    for (int j = 1; j < NC; ++j) m = fmaxf(m, lg[j]);
    float s = 0.0f;
#pragma unroll
    for (int j = 0; j < NC; ++j) s += __expf(lg[j] - m);
    float lse = m + __logf(s);
#pragma unroll
    for (int j = 0; j < NC; ++j) out[b * NC + j] = lg[j] - lse;
}

// ---------------------------------------------------------------------------
// Orchestration
// ---------------------------------------------------------------------------
extern "C" void kernel_launch(void* const* d_in, const int* in_sizes, int n_in,
                              void* d_out, int out_size, void* d_ws, size_t ws_size,
                              hipStream_t stream)
{
    const float* x          = (const float*)d_in[0];
    const float* W_conv     = (const float*)d_in[1];
    const float* b_conv     = (const float*)d_in[2];
    const float* bn1_g      = (const float*)d_in[3];
    const float* bn1_b      = (const float*)d_in[4];
    const float* in_proj_w  = (const float*)d_in[5];   // [768,256]
    const float* in_proj_b  = (const float*)d_in[6];
    const float* out_proj_w = (const float*)d_in[7];   // [256,256]
    const float* out_proj_b = (const float*)d_in[8];
    const float* bn2_g      = (const float*)d_in[9];
    const float* bn2_b      = (const float*)d_in[10];
    const float* mlp_w1     = (const float*)d_in[11];  // [256,512]
    const float* mlp_b1     = (const float*)d_in[12];
    const float* mlp_w2     = (const float*)d_in[13];  // [512,256]
    const float* mlp_b2     = (const float*)d_in[14];
    const float* bn3_g      = (const float*)d_in[15];
    const float* bn3_b      = (const float*)d_in[16];
    const float* fc_w       = (const float*)d_in[17];  // [256,10]
    const float* fc_b       = (const float*)d_in[18];
    const int*   edge_index = (const int*)d_in[19];    // [2,E] flat
    (void)in_sizes; (void)n_in; (void)out_size; (void)ws_size;

    float* ws = (float*)d_ws;
    const size_t NC_ = (size_t)NN * CC;                // 8.39M floats
    float* bufA   = ws;                                 // N*768  (qkv -> ff1)
    float* bufB   = bufA + (size_t)NN * 3 * CC;         // N*256  (h_lin -> ff2)
    float* bufC   = bufB + NC_;                         // N*256  (agg -> attn_out -> out3)
    float* bufD   = bufC + NC_;                         // N*256  (h1)
    float* bufE   = bufD + NC_;                         // N*256  (attn_proj -> out)
    float* dinv   = bufE + NC_;                         // N
    float* stats  = dinv + NN;                          // 512
    float* pooled = stats + 2 * CC;                     // B*C

    // ---- GCN: h = x @ W_conv ----
    gemm_wmma<false, false, false><<<dim3(NN / 128, CC / 16), 256, 0, stream>>>(
        x, W_conv, nullptr, bufB, NN, CC, CC);

    // degrees (with self-loop), dinv = rsqrt(deg)
    deg_init<<<NN / 256, 256, 0, stream>>>(dinv, NN);
    deg_accum<<<EE / 256, 256, 0, stream>>>(edge_index + EE, dinv, EE);
    deg_rsqrt<<<NN / 256, 256, 0, stream>>>(dinv, NN);

    // aggregate: self-loop init then edge scatter-add
    agg_init<<<(int)(NC_ / 256), 256, 0, stream>>>(bufB, dinv, bufC, (int)NC_);
    agg_edges<<<EE / 4, 256, 0, stream>>>(edge_index, bufB, dinv, bufC, EE);

    // h1 = BN1(agg + b_conv + x)
    (void)hipMemsetAsync(stats, 0, 2 * CC * sizeof(float), stream);
    bn_stats<<<256, 256, 0, stream>>>(bufC, x, b_conv, stats);
    bn_apply<<<(int)(NC_ / 256), 256, 0, stream>>>(bufC, x, b_conv, stats,
                                                   bn1_g, bn1_b, nullptr, bufD, (int)NC_);

    // ---- attention branch ----
    // qkv = x @ in_proj_w^T + in_proj_b    [N,768]
    gemm_wmma<true, true, false><<<dim3(NN / 128, (3 * CC) / 16), 256, 0, stream>>>(
        x, in_proj_w, in_proj_b, bufA, NN, CC, 3 * CC);

    attn_kernel<<<dim3(PER / 64, HH, BB), 256, 0, stream>>>(bufA, bufC);

    // o @ out_proj_w^T + out_proj_b
    gemm_wmma<true, true, false><<<dim3(NN / 128, CC / 16), 256, 0, stream>>>(
        bufC, out_proj_w, out_proj_b, bufE, NN, CC, CC);

    // out = h1 + BN2(attn_proj + x)   (in-place into bufE)
    (void)hipMemsetAsync(stats, 0, 2 * CC * sizeof(float), stream);
    bn_stats<<<256, 256, 0, stream>>>(bufE, x, nullptr, stats);
    bn_apply<<<(int)(NC_ / 256), 256, 0, stream>>>(bufE, x, nullptr, stats,
                                                   bn2_g, bn2_b, bufD, bufE, (int)NC_);

    // ---- FFN ----
    gemm_wmma<false, true, true><<<dim3(NN / 128, (2 * CC) / 16), 256, 0, stream>>>(
        bufE, mlp_w1, mlp_b1, bufA, NN, CC, 2 * CC);
    gemm_wmma<false, true, false><<<dim3(NN / 128, CC / 16), 256, 0, stream>>>(
        bufA, mlp_w2, mlp_b2, bufB, NN, 2 * CC, CC);

    // out3 = BN3(out + ff)
    (void)hipMemsetAsync(stats, 0, 2 * CC * sizeof(float), stream);
    bn_stats<<<256, 256, 0, stream>>>(bufE, bufB, nullptr, stats);
    bn_apply<<<(int)(NC_ / 256), 256, 0, stream>>>(bufE, bufB, nullptr, stats,
                                                   bn3_g, bn3_b, nullptr, bufC, (int)NC_);

    // ---- pool + classify ----
    pool_max<<<BB, 256, 0, stream>>>(bufC, pooled);
    classify<<<1, 128, 0, stream>>>(pooled, fc_w, fc_b, (float*)d_out);
}
